// TurbulenceRobustLoss_61100204753515
// MI455X (gfx1250) — compile-verified, matrix operand order checked by reading, not run on hardware
//
#include <hip/hip_runtime.h>
#include <hip/hip_bf16.h>
#include <math.h>

// ---------------------------------------------------------------------------
// TurbulenceRobustLoss for MI455X (gfx1250, wave32, WMMA)
//   B = 8192 rows, D = 256 features, rfft bins = 129, T = 0.07
//   out = [total, spatial, frequency, contrastive]  (4 x f32)
// ---------------------------------------------------------------------------

#define BROWS 8192
#define DFEAT 256
#define NBINS 129
#define INV_T 14.2857142857f   // 1/0.07

typedef __attribute__((ext_vector_type(16))) __bf16 v16bf;
typedef __attribute__((ext_vector_type(8)))  float  v8f;

union Frag {
    v16bf v;
    uint4 q[2];
};

// ---------------------------------------------------------------------------
// K1: L2-normalize rows (f32 -> bf16) + per-row spatial-MSE partial
// ---------------------------------------------------------------------------
__global__ __launch_bounds__(256)
void k_norm_spatial(const float* __restrict__ clean,
                    const float* __restrict__ turb,
                    __bf16* __restrict__ cnbf,
                    __bf16* __restrict__ tnbf,
                    float* __restrict__ spat_part)
{
    __shared__ float sC[256], sT[256], sD[256];
    const int row = blockIdx.x;
    const int d   = threadIdx.x;
    const float c = clean[(size_t)row * DFEAT + d];
    const float t = turb [(size_t)row * DFEAT + d];
    const float diff = c - t;
    sC[d] = c * c;
    sT[d] = t * t;
    sD[d] = diff * diff;
    __syncthreads();
    for (int s = 128; s > 0; s >>= 1) {
        if (d < s) {
            sC[d] += sC[d + s];
            sT[d] += sT[d + s];
            sD[d] += sD[d + s];
        }
        __syncthreads();
    }
    const float invc = 1.0f / fmaxf(sqrtf(sC[0]), 1e-12f);
    const float invt = 1.0f / fmaxf(sqrtf(sT[0]), 1e-12f);
    cnbf[(size_t)row * DFEAT + d] = (__bf16)(c * invc);
    tnbf[(size_t)row * DFEAT + d] = (__bf16)(t * invt);
    if (d == 0) spat_part[row] = sD[0];
}

// ---------------------------------------------------------------------------
// K2: per-row rFFT-magnitude MSE partial (DFT via twiddle recurrence)
// ---------------------------------------------------------------------------
__global__ __launch_bounds__(256)
void k_freq(const float* __restrict__ clean,
            const float* __restrict__ turb,
            float* __restrict__ freq_part)
{
    __shared__ float sc[256], st[256], red[256];
    const int row = blockIdx.x;
    const int tid = threadIdx.x;
    sc[tid] = clean[(size_t)row * DFEAT + tid];
    st[tid] = turb [(size_t)row * DFEAT + tid];
    __syncthreads();

    float val = 0.0f;
    if (tid < NBINS) {
        const float theta = -6.28318530718f * (float)tid / (float)DFEAT;
        float sn, cs;
        __sincosf(theta, &sn, &cs);
        float cr = 1.0f, ci = 0.0f;               // running twiddle e^{j*theta*d}
        float rc = 0.0f, ic = 0.0f, rt = 0.0f, it = 0.0f;
        #pragma unroll 4
        for (int d = 0; d < DFEAT; ++d) {
            const float xc = sc[d];
            const float xt = st[d];
            rc += xc * cr;  ic += xc * ci;
            rt += xt * cr;  it += xt * ci;
            const float nr = cr * cs - ci * sn;   // rotate twiddle
            ci = cr * sn + ci * cs;
            cr = nr;
        }
        const float mc = sqrtf(rc * rc + ic * ic);
        const float mt = sqrtf(rt * rt + it * it);
        const float dm = mc - mt;
        val = dm * dm;
    }
    red[tid] = val;
    __syncthreads();
    for (int s = 128; s > 0; s >>= 1) {
        if (tid < s) red[tid] += red[tid + s];
        __syncthreads();
    }
    if (tid == 0) freq_part[row] = red[0];
}

// ---------------------------------------------------------------------------
// K3: fused  exp(cn . tn^T / T)  with masked row sums, bf16 WMMA, f32 acc.
//   Block tile 64(M) x 128(N), 8 waves as 2x4, each wave a 32x32 patch.
//   Writes all_part[jb*B + i], pos_part[jb*B + i]  (each exactly once).
// ---------------------------------------------------------------------------
__global__ __launch_bounds__(256)
void k_sim(const __bf16* __restrict__ cnbf,
           const __bf16* __restrict__ tnbf,
           const int*    __restrict__ sids,
           float* __restrict__ all_part,
           float* __restrict__ pos_part)
{
    __shared__ int   s_rowids[64];
    __shared__ int   s_colids[128];
    __shared__ float s_all[64][4];
    __shared__ float s_pos[64][4];

    const int tid  = threadIdx.x;
    const int wave = tid >> 5;
    const int lane = tid & 31;
    const int wr   = wave >> 2;          // 0..1  (row group of 32)
    const int wc   = wave & 3;           // 0..3  (col group of 32)
    const int jb   = blockIdx.x;         // 64 column blocks of 128
    const int ib   = blockIdx.y;         // 128 row blocks of 64
    const int m0   = ib * 64;
    const int n0   = jb * 128;

    if (tid < 64)  s_rowids[tid] = sids[m0 + tid];
    if (tid < 128) s_colids[tid] = sids[n0 + tid];
    __syncthreads();

    const int half = lane >> 4;          // K-half select per CDNA5 16-bit layout
    const int l15  = lane & 15;

    v8f acc[2][2];
    #pragma unroll
    for (int mi = 0; mi < 2; ++mi)
        #pragma unroll
        for (int ni = 0; ni < 2; ++ni)
            #pragma unroll
            for (int e = 0; e < 8; ++e)
                acc[mi][ni][e] = 0.0f;

    // K loop: 256 = 8 steps of K=32
    for (int k0 = 0; k0 < DFEAT; k0 += 32) {
        Frag a[2], b[2];
        #pragma unroll
        for (int mi = 0; mi < 2; ++mi) {
            // A layout (16-bit, 16x32): lane l15 = M row; e0..7 -> K = half*8 + e,
            // e8..15 -> K = 16 + half*8 + e   => two contiguous 16B loads
            const __bf16* p = cnbf +
                (size_t)(m0 + wr * 32 + mi * 16 + l15) * DFEAT + k0 + half * 8;
            a[mi].q[0] = *(const uint4*)(p);
            a[mi].q[1] = *(const uint4*)(p + 16);
        }
        #pragma unroll
        for (int ni = 0; ni < 2; ++ni) {
            // B (32x16) mirrored layout: lane l15 = N col, same K striping,
            // and B[k][n] = tn[n][k]  => identical gather from row-major tn
            const __bf16* p = tnbf +
                (size_t)(n0 + wc * 32 + ni * 16 + l15) * DFEAT + k0 + half * 8;
            b[ni].q[0] = *(const uint4*)(p);
            b[ni].q[1] = *(const uint4*)(p + 16);
        }
        #pragma unroll
        for (int mi = 0; mi < 2; ++mi)
            #pragma unroll
            for (int ni = 0; ni < 2; ++ni)
                acc[mi][ni] = __builtin_amdgcn_wmma_f32_16x16x32_bf16(
                    false, a[mi].v, false, b[ni].v,
                    (short)0, acc[mi][ni], false, false);
    }

    // Epilogue: exp(sim/T), positive mask, per-row sums.
    // C/D layout: VGPR v, lane L -> (M = v + 8*(L>=16), N = L&15)
    #pragma unroll
    for (int mi = 0; mi < 2; ++mi) {
        #pragma unroll
        for (int v = 0; v < 8; ++v) {
            const int rlocal = wr * 32 + mi * 16 + v + 8 * half;
            const int srow   = s_rowids[rlocal];
            float esum = 0.0f, psum = 0.0f;
            #pragma unroll
            for (int ni = 0; ni < 2; ++ni) {
                const float e    = __expf(acc[mi][ni][v] * INV_T);
                const int  scol  = s_colids[wc * 32 + ni * 16 + l15];
                esum += e;
                psum += (srow == scol) ? e : 0.0f;
            }
            // reduce across the 16 lanes holding one output row
            #pragma unroll
            for (int m = 1; m < 16; m <<= 1) {
                esum += __shfl_xor(esum, m, 32);
                psum += __shfl_xor(psum, m, 32);
            }
            if (l15 == 0) {
                s_all[rlocal][wc] = esum;
                s_pos[rlocal][wc] = psum;
            }
        }
    }
    __syncthreads();

    if (tid < 64) {
        const float a4 = s_all[tid][0] + s_all[tid][1] + s_all[tid][2] + s_all[tid][3];
        const float p4 = s_pos[tid][0] + s_pos[tid][1] + s_pos[tid][2] + s_pos[tid][3];
        all_part[(size_t)jb * BROWS + m0 + tid] = a4;
        pos_part[(size_t)jb * BROWS + m0 + tid] = p4;
    }
}

// ---------------------------------------------------------------------------
// K4: per-row InfoNCE term, fixed-order sum over the 64 column blocks
// ---------------------------------------------------------------------------
__global__ __launch_bounds__(256)
void k_rowloss(const float* __restrict__ all_part,
               const float* __restrict__ pos_part,
               float* __restrict__ contr_part)
{
    __shared__ float red[256];
    const int tid = threadIdx.x;
    const int i   = blockIdx.x * 256 + tid;
    float a = 0.0f, p = 0.0f;
    for (int jb = 0; jb < 64; ++jb) {
        a += all_part[(size_t)jb * BROWS + i];
        p += pos_part[(size_t)jb * BROWS + i];
    }
    red[tid] = -logf(p / (a + 1e-8f));
    __syncthreads();
    for (int s = 128; s > 0; s >>= 1) {
        if (tid < s) red[tid] += red[tid + s];
        __syncthreads();
    }
    if (tid == 0) contr_part[blockIdx.x] = red[0];
}

// ---------------------------------------------------------------------------
// K5: final deterministic combine
// ---------------------------------------------------------------------------
__global__ __launch_bounds__(256)
void k_final(const float* __restrict__ spat_part,
             const float* __restrict__ freq_part,
             const float* __restrict__ contr_part,
             float* __restrict__ out)
{
    __shared__ float r1[256], r2[256], r3[256];
    const int t = threadIdx.x;
    float s = 0.0f, f = 0.0f, c = 0.0f;
    for (int i = t; i < BROWS; i += 256) {
        s += spat_part[i];
        f += freq_part[i];
    }
    if (t < 32) c = contr_part[t];
    r1[t] = s; r2[t] = f; r3[t] = c;
    __syncthreads();
    for (int k = 128; k > 0; k >>= 1) {
        if (t < k) { r1[t] += r1[t + k]; r2[t] += r2[t + k]; r3[t] += r3[t + k]; }
        __syncthreads();
    }
    if (t == 0) {
        const float spatial     = r1[0] / ((float)BROWS * (float)DFEAT);
        const float frequency   = r2[0] / ((float)BROWS * (float)NBINS);
        const float contrastive = r3[0] / (float)BROWS;
        out[0] = spatial + frequency + 0.5f * contrastive;
        out[1] = spatial;
        out[2] = frequency;
        out[3] = contrastive;
    }
}

// ---------------------------------------------------------------------------
extern "C" void kernel_launch(void* const* d_in, const int* in_sizes, int n_in,
                              void* d_out, int out_size, void* d_ws, size_t ws_size,
                              hipStream_t stream)
{
    const float* clean = (const float*)d_in[0];
    const float* turb  = (const float*)d_in[1];
    const int*   sids  = (const int*)d_in[2];
    float*       out   = (float*)d_out;

    char* ws = (char*)d_ws;
    __bf16* cnbf      = (__bf16*)(ws);                               // 4 MB
    __bf16* tnbf      = (__bf16*)(ws + (size_t)4 * 1024 * 1024);     // 4 MB
    float*  all_part  = (float*)(ws + (size_t)8  * 1024 * 1024);     // 2 MB (64 x 8192)
    float*  pos_part  = (float*)(ws + (size_t)10 * 1024 * 1024);     // 2 MB
    float*  spat_part = (float*)(ws + (size_t)12 * 1024 * 1024);     // 32 KB
    float*  freq_part = (float*)(ws + (size_t)12 * 1024 * 1024 + 32768);
    float*  contr_part= (float*)(ws + (size_t)12 * 1024 * 1024 + 65536);

    k_norm_spatial<<<BROWS, 256, 0, stream>>>(clean, turb, cnbf, tnbf, spat_part);
    k_freq        <<<BROWS, 256, 0, stream>>>(clean, turb, freq_part);
    k_sim         <<<dim3(64, 128), 256, 0, stream>>>(cnbf, tnbf, sids, all_part, pos_part);
    k_rowloss     <<<32, 256, 0, stream>>>(all_part, pos_part, contr_part);
    k_final       <<<1, 256, 0, stream>>>(spat_part, freq_part, contr_part, out);
}
